// RhythmPatternExtractor_82222853915142
// MI455X (gfx1250) — matrix operand order repaired and checked by reading, not compiled, not run on hardware
//
#include <hip/hip_runtime.h>
#include <hip/hip_bf16.h>
#include <math.h>

// ---------------------------------------------------------------------------
// RhythmPatternExtractor for MI455X (gfx1250, wave32, WMMA f16 16x16x32)
//
// Pipeline: binarized conv1d (implicit GEMM, M=64 K=3456 N=B*T) -> fused
// BN+ELU+spike -> QKV projections -> flash-attention (online softmax, never
// materializes the 200MB score tensor) -> output projection + residual.
// All matmuls use V_WMMA_F32_16X16X32_F16. Every WMMA operand fragment is
// fetched with contiguous 16B vector loads (global or LDS); strided operands
// are staged transposed in LDS so B-fragments are two ds_load_b128.
// ---------------------------------------------------------------------------

typedef _Float16 h16;
typedef __attribute__((ext_vector_type(16))) _Float16 v16h;
typedef __attribute__((ext_vector_type(8)))  _Float16 v8h;
typedef __attribute__((ext_vector_type(8)))  float    v8f;

#define B_N   128
#define C_IN  128
#define T_IN  2500
#define C_O   64
#define K_W   27
#define STR   4
#define PADL  13
#define T_O   625
#define T_OP  640          // padded T (multiple of 64); pad region zero-filled
#define KD    (C_IN * K_W) // 3456
#define KSTEPS (KD / 32)   // 108
#define CONV_NT 32
#define XSPAN 152          // (CONV_NT-1)*STR + K_W = 151, padded
#define CPAD  136          // conv x-slab row stride (halves); 16B aligned rows
#define TPAD  72           // transposed [t][c] tile row stride (halves)

union U16 { v16h v; v8h h[2]; };

static __device__ __forceinline__ v8f wmma_f32_f16(v16h a, v16h b, v8f c) {
  return __builtin_amdgcn_wmma_f32_16x16x32_f16(false, a, false, b, (short)0, c,
                                                false, false);
}

// A fragment (16x32 f16) from row-major A[m][k]; two contiguous 16B loads.
// Lane L: m = mBase + (L&15), g = L>>4; half j -> k = kBase+16g+j
// (contiguous order, matching the B-fragment loaders).
static __device__ __forceinline__ v16h load_a(const h16* A, int ld, int mBase,
                                              int kBase) {
  int lane = threadIdx.x & 31;
  int m = mBase + (lane & 15);
  int g = lane >> 4;
  const h16* p = A + m * ld + kBase + 16 * g;
  U16 u;
  u.h[0] = *(const v8h*)p;
  u.h[1] = *(const v8h*)(p + 8);
  return u.v;
}

// B fragment (32x16 f16) where the source is stored n-major with contiguous
// k: Bt[n][k] (ld = row stride). Lane L: n = nBase + (L&15), g = L>>4;
// half j -> k = kBase + 16g + j -> 16 contiguous halves = two 16B loads.
static __device__ __forceinline__ v16h load_bT(const h16* Bt, int ld, int kBase,
                                               int nBase) {
  int lane = threadIdx.x & 31;
  int n = nBase + (lane & 15);
  int g = lane >> 4;
  const h16* p = Bt + n * ld + kBase + 16 * g;
  U16 u;
  u.h[0] = *(const v8h*)p;
  u.h[1] = *(const v8h*)(p + 8);
  return u.v;
}

// Conv B fragment from the position-major x-slab xs[p][c] (row stride CPAD).
// kk = tap*128 + c, base = ks*32 + 16g is 16-aligned so the 16 halves have
// constant tap and consecutive c -> 32 contiguous bytes in LDS.
static __device__ __forceinline__ v16h load_b_conv(const h16* xs, int ks,
                                                   int ltBase) {
  int lane = threadIdx.x & 31;
  int g = lane >> 4, hn = lane & 15;
  int base = ks * 32 + 16 * g;
  int p = 4 * (ltBase + hn) + (base >> 7);
  const h16* q = xs + p * CPAD + (base & 127);
  U16 u;
  u.h[0] = *(const v8h*)q;
  u.h[1] = *(const v8h*)(q + 8);
  return u.v;
}

// --------------------------- kernel A: prep --------------------------------
__global__ void __launch_bounds__(256) prep_weights(
    const float* __restrict__ w, const float* __restrict__ conv_b,
    const float* __restrict__ gamma, const float* __restrict__ beta,
    const float* __restrict__ mean, const float* __restrict__ var,
    h16* __restrict__ wp, float* __restrict__ scale, float* __restrict__ shift) {
  int o = blockIdx.x;
  int tid = threadIdx.x;
  __shared__ float red[256];
  __shared__ float s_alpha;
  const float* wo_ = w + o * KD;
  float s = 0.f;
  for (int i = tid; i < KD; i += 256) s += fabsf(wo_[i]);
  red[tid] = s;
  __syncthreads();
  for (int st = 128; st > 0; st >>= 1) {
    if (tid < st) red[tid] += red[tid + st];
    __syncthreads();
  }
  if (tid == 0) {
    float alpha = red[0] / (float)KD;
    s_alpha = alpha;
    float inv = rsqrtf(var[o] + 1e-5f);
    float sc = gamma[o] * inv;
    scale[o] = sc;
    shift[o] = beta[o] + (conv_b[o] - mean[o]) * sc;
  }
  __syncthreads();
  float a = s_alpha;
  for (int i = tid; i < KD; i += 256) {
    int k = i >> 7, c = i & 127;          // kk = tap*128 + c
    float v = wo_[c * K_W + k];           // input layout (o, c, k)
    float q = (v > 0.f) ? a : (v < 0.f ? -a : 0.f);
    wp[o * KD + i] = (h16)q;
  }
}

// --------------------- kernel F: linear weights -> f16 ----------------------
__global__ void __launch_bounds__(256) prep_lin(
    const float* __restrict__ wq, const float* __restrict__ wk,
    const float* __restrict__ wv, const float* __restrict__ wo,
    h16* __restrict__ wq16, h16* __restrict__ wk16, h16* __restrict__ wv16,
    h16* __restrict__ wo16) {
  int i = blockIdx.x * 256 + threadIdx.x;  // 0..16383
  int which = i >> 12, r = i & 4095;
  if (which == 0) wq16[r] = (h16)(wq[r] * 0.125f);  // fold 1/sqrt(64)
  else if (which == 1) wk16[r] = (h16)(wk[r]);
  else if (which == 2) wv16[r] = (h16)(wv[r]);
  else wo16[r] = (h16)(wo[r]);
}

// ------------------ kernel B: conv (implicit GEMM) + BN + ELU ---------------
__global__ void __launch_bounds__(128) conv_bn_elu(
    const float* __restrict__ x, const h16* __restrict__ wp,
    const float* __restrict__ scale, const float* __restrict__ shift,
    float* __restrict__ y32, h16* __restrict__ y16, h16* __restrict__ s16) {
  __shared__ __align__(16) h16 xs[XSPAN * CPAD];  // 41.3 KB, [p][c] layout
  int tile = blockIdx.x;
  int b = blockIdx.y;
  int t0 = tile * CONV_NT;
  int p0 = t0 * STR - PADL;
  int tid = threadIdx.x;
  const float* xb = x + (size_t)b * C_IN * T_IN;
  // stage transposed: p fastest -> coalesced global reads, conflict-free ds
  for (int i = tid; i < C_IN * XSPAN; i += 128) {
    int c = i / XSPAN, p = i - c * XSPAN;
    int gp = p0 + p;
    float v = 0.f;
    if (gp >= 0 && gp < T_IN) v = xb[c * T_IN + gp];
    xs[p * CPAD + c] = (h16)v;
  }
  __syncthreads();
  int wv = tid >> 5, lane = tid & 31, g = lane >> 4, hn = lane & 15;
  v8f acc0 = {}, acc1 = {};
#pragma unroll 4
  for (int ks = 0; ks < KSTEPS; ++ks) {
    v16h a = load_a(wp, KD, wv * 16, ks * 32);
    if (ks + 1 < KSTEPS)  // stream next weight chunk (global_prefetch_b8)
      __builtin_prefetch(wp + (wv * 16 + hn) * KD + (ks + 1) * 32, 0, 1);
    v16h b0 = load_b_conv(xs, ks, 0);
    v16h b1 = load_b_conv(xs, ks, 16);
    acc0 = wmma_f32_f16(a, b0, acc0);
    acc1 = wmma_f32_f16(a, b1, acc1);
  }
#pragma unroll
  for (int r = 0; r < 8; ++r) {
    int o = wv * 16 + 8 * g + r;
    float sc = scale[o], sh = shift[o];
#pragma unroll
    for (int nt = 0; nt < 2; ++nt) {
      int t = t0 + nt * 16 + hn;
      float v = (nt ? acc1[r] : acc0[r]) * sc + sh;
      float e = v > 0.f ? v : (__expf(v) - 1.f);  // ELU
      float sp = v > 0.f ? 1.f : 0.f;             // spike (forward = hard)
      if (t >= T_O) { e = 0.f; sp = 0.f; }        // zero-fill pad region
      size_t idx = ((size_t)b * C_O + o) * T_OP + t;
      y32[idx] = e;
      y16[idx] = (h16)e;
      s16[idx] = (h16)sp;
    }
  }
}

// ------------------------- kernel C: QKV projections ------------------------
__global__ void __launch_bounds__(128) qkv_proj(
    const h16* __restrict__ s16, const h16* __restrict__ y16,
    const h16* __restrict__ wq16, const h16* __restrict__ wk16,
    const h16* __restrict__ wv16, h16* __restrict__ qT,
    h16* __restrict__ kbuf, h16* __restrict__ vbuf) {
  __shared__ __align__(16) h16 sl[64 * TPAD];  // transposed [t][c]
  __shared__ __align__(16) h16 yl[64 * TPAD];
  int t0 = blockIdx.x * 64;
  int b = blockIdx.y;
  int tid = threadIdx.x;
  for (int i = tid; i < C_O * 64; i += 128) {
    int c = i >> 6, tt = i & 63;  // tt fastest: coalesced reads
    size_t idx = ((size_t)b * C_O + c) * T_OP + t0 + tt;
    sl[tt * TPAD + c] = s16[idx];
    yl[tt * TPAD + c] = y16[idx];
  }
  __syncthreads();
  int wv = tid >> 5, lane = tid & 31, g = lane >> 4, hn = lane & 15;
  int nBase = wv * 16;
  v8f accq[4] = {}, acck[4] = {}, accv[4] = {};
#pragma unroll
  for (int kt = 0; kt < 2; ++kt) {
    v16h bS = load_bT(sl, TPAD, kt * 32, nBase);
    v16h bY = load_bT(yl, TPAD, kt * 32, nBase);
#pragma unroll
    for (int mt = 0; mt < 4; ++mt) {
      v16h aQ = load_a(wq16, 64, mt * 16, kt * 32);
      v16h aK = load_a(wk16, 64, mt * 16, kt * 32);
      v16h aV = load_a(wv16, 64, mt * 16, kt * 32);
      accq[mt] = wmma_f32_f16(aQ, bS, accq[mt]);
      acck[mt] = wmma_f32_f16(aK, bS, acck[mt]);
      accv[mt] = wmma_f32_f16(aV, bY, accv[mt]);
    }
  }
  int t = t0 + nBase + hn;
  bool valid = (t < T_O);
#pragma unroll
  for (int mt = 0; mt < 4; ++mt) {
#pragma unroll
    for (int r = 0; r < 8; ++r) {
      int o = mt * 16 + 8 * g + r;
      qT[((size_t)b * T_OP + t) * C_O + o]   = (h16)(valid ? accq[mt][r] : 0.f);
      kbuf[((size_t)b * C_O + o) * T_OP + t] = (h16)(valid ? acck[mt][r] : 0.f);
      vbuf[((size_t)b * C_O + o) * T_OP + t] = (h16)(valid ? accv[mt][r] : 0.f);
    }
  }
}

// --------------------- kernel D: flash attention + ctx ----------------------
__global__ void __launch_bounds__(128) attention(
    const h16* __restrict__ qT, const h16* __restrict__ kbuf,
    const h16* __restrict__ vbuf, h16* __restrict__ ctx) {
  __shared__ __align__(16) h16 kl[64 * TPAD];      // K tile transposed [s][c]
  __shared__ __align__(16) h16 pl[4 * 16 * 64];    // per-wave P tiles
  int qt0 = blockIdx.x * 64;
  int b = blockIdx.y;
  int tid = threadIdx.x;
  int wv = tid >> 5, lane = tid & 31, g = lane >> 4, hn = lane & 15;
  const h16* qTb = qT + ((size_t)b * T_OP + qt0) * C_O;
  v16h aQ0 = load_a(qTb, C_O, wv * 16, 0);
  v16h aQ1 = load_a(qTb, C_O, wv * 16, 32);
  float rowM[8], rowL[8];
#pragma unroll
  for (int r = 0; r < 8; ++r) { rowM[r] = -1e30f; rowL[r] = 0.f; }
  v8f oacc[4] = {};
  h16* plw = pl + wv * 16 * 64;
  const h16* kb = kbuf + (size_t)b * C_O * T_OP;
  const h16* vb = vbuf + (size_t)b * C_O * T_OP;
  for (int sb = 0; sb < T_OP; sb += 64) {
    __syncthreads();
    for (int i = tid; i < C_O * 64; i += 128) {
      int c = i >> 6, ss = i & 63;  // ss fastest: coalesced
      int s = sb + ss;
      kl[ss * TPAD + c] = (s < T_O) ? kb[(size_t)c * T_OP + s] : (h16)0.f;
    }
    __syncthreads();
    v8f sc[4] = {};
#pragma unroll
    for (int nt = 0; nt < 4; ++nt) {
      v16h b0 = load_bT(kl, TPAD, 0, nt * 16);
      v16h b1 = load_bT(kl, TPAD, 32, nt * 16);
      sc[nt] = wmma_f32_f16(aQ0, b0, sc[nt]);
      sc[nt] = wmma_f32_f16(aQ1, b1, sc[nt]);
    }
    // online softmax: rows live as (g,r) per lane, cols across 16 lanes
#pragma unroll
    for (int r = 0; r < 8; ++r) {
      float mx = -1e30f;
#pragma unroll
      for (int nt = 0; nt < 4; ++nt) {
        int s = sb + nt * 16 + hn;
        float v = (s < T_O) ? sc[nt][r] : -1e30f;
        sc[nt][r] = v;
        mx = fmaxf(mx, v);
      }
#pragma unroll
      for (int off = 1; off < 16; off <<= 1) mx = fmaxf(mx, __shfl_xor(mx, off));
      float nm = fmaxf(rowM[r], mx);
      float corr = __expf(rowM[r] - nm);
      float ps = 0.f;
#pragma unroll
      for (int nt = 0; nt < 4; ++nt) {
        float p = __expf(sc[nt][r] - nm);
        sc[nt][r] = p;
        ps += p;
      }
#pragma unroll
      for (int off = 1; off < 16; off <<= 1) ps += __shfl_xor(ps, off);
      rowL[r] = rowL[r] * corr + ps;
      rowM[r] = nm;
#pragma unroll
      for (int ct = 0; ct < 4; ++ct) oacc[ct][r] *= corr;
    }
    // P -> LDS (per-wave region) for use as A-matrix of ctx GEMM
#pragma unroll
    for (int r = 0; r < 8; ++r)
#pragma unroll
      for (int nt = 0; nt < 4; ++nt)
        plw[(8 * g + r) * 64 + nt * 16 + hn] = (h16)sc[nt][r];
    __syncthreads();
    v16h pA0 = load_a(plw, 64, 0, 0);
    v16h pA1 = load_a(plw, 64, 0, 32);
    const h16* vbs = vb + sb;  // v is s-contiguous -> vector B-frag loads
#pragma unroll
    for (int ct = 0; ct < 4; ++ct) {
      v16h bv0 = load_bT(vbs, T_OP, 0, ct * 16);
      v16h bv1 = load_bT(vbs, T_OP, 32, ct * 16);
      oacc[ct] = wmma_f32_f16(pA0, bv0, oacc[ct]);
      oacc[ct] = wmma_f32_f16(pA1, bv1, oacc[ct]);
    }
  }
#pragma unroll
  for (int r = 0; r < 8; ++r) {
    int t = qt0 + wv * 16 + 8 * g + r;
    float inv = 1.f / rowL[r];
#pragma unroll
    for (int ct = 0; ct < 4; ++ct) {
      int c = ct * 16 + hn;
      float val = (t < T_O) ? oacc[ct][r] * inv : 0.f;
      ctx[((size_t)b * C_O + c) * T_OP + t] = (h16)val;
    }
  }
}

// ----------------- kernel E: output projection + residual -------------------
__global__ void __launch_bounds__(128) out_proj(
    const h16* __restrict__ ctx, const h16* __restrict__ wo16,
    const float* __restrict__ y32, float* __restrict__ out) {
  __shared__ __align__(16) h16 cl[64 * TPAD];  // transposed [t][c]
  int t0 = blockIdx.x * 64;
  int b = blockIdx.y;
  int tid = threadIdx.x;
  for (int i = tid; i < C_O * 64; i += 128) {
    int c = i >> 6, tt = i & 63;
    cl[tt * TPAD + c] = ctx[((size_t)b * C_O + c) * T_OP + t0 + tt];
  }
  __syncthreads();
  int wv = tid >> 5, lane = tid & 31, g = lane >> 4, hn = lane & 15;
  int nBase = wv * 16;
  v8f acc[4] = {};
#pragma unroll
  for (int kt = 0; kt < 2; ++kt) {
    v16h bC = load_bT(cl, TPAD, kt * 32, nBase);
#pragma unroll
    for (int mt = 0; mt < 4; ++mt) {
      v16h aO = load_a(wo16, 64, mt * 16, kt * 32);
      acc[mt] = wmma_f32_f16(aO, bC, acc[mt]);
    }
  }
  int t = t0 + nBase + hn;
  if (t < T_O) {
#pragma unroll
    for (int mt = 0; mt < 4; ++mt)
#pragma unroll
      for (int r = 0; r < 8; ++r) {
        int o = mt * 16 + 8 * g + r;
        float yv = y32[((size_t)b * C_O + o) * T_OP + t];
        out[((size_t)b * C_O + o) * T_O + t] = yv + acc[mt][r];
      }
  }
}

// ---------------------------------------------------------------------------
extern "C" void kernel_launch(void* const* d_in, const int* in_sizes, int n_in,
                              void* d_out, int out_size, void* d_ws,
                              size_t ws_size, hipStream_t stream) {
  const float* x      = (const float*)d_in[0];
  const float* conv_w = (const float*)d_in[1];
  const float* conv_b = (const float*)d_in[2];
  const float* bn_g   = (const float*)d_in[3];
  const float* bn_b   = (const float*)d_in[4];
  const float* bn_m   = (const float*)d_in[5];
  const float* bn_v   = (const float*)d_in[6];
  const float* wq     = (const float*)d_in[7];
  const float* wk     = (const float*)d_in[8];
  const float* wvv    = (const float*)d_in[9];
  const float* wo     = (const float*)d_in[10];
  (void)in_sizes; (void)n_in; (void)out_size; (void)ws_size;

  char* ws = (char*)d_ws;
  size_t off = 0;
  auto alloc = [&](size_t bytes) {
    size_t o = off;
    off = (off + bytes + 255) & ~(size_t)255;
    return o;
  };
  h16*   wp    = (h16*)(ws + alloc((size_t)C_O * KD * 2));
  float* scale = (float*)(ws + alloc(C_O * 4));
  float* shift = (float*)(ws + alloc(C_O * 4));
  h16* wq16 = (h16*)(ws + alloc(4096 * 2));
  h16* wk16 = (h16*)(ws + alloc(4096 * 2));
  h16* wv16 = (h16*)(ws + alloc(4096 * 2));
  h16* wo16 = (h16*)(ws + alloc(4096 * 2));
  size_t plane = (size_t)B_N * C_O * T_OP;           // padded (b, c, t)
  float* y32 = (float*)(ws + alloc(plane * 4));
  h16* y16 = (h16*)(ws + alloc(plane * 2));
  h16* s16 = (h16*)(ws + alloc(plane * 2));
  h16* qT  = (h16*)(ws + alloc((size_t)B_N * T_OP * C_O * 2));
  h16* kb  = (h16*)(ws + alloc(plane * 2));
  h16* vb  = (h16*)(ws + alloc(plane * 2));
  h16* cx  = (h16*)(ws + alloc(plane * 2));
  // total workspace ~84.4 MB

  prep_weights<<<dim3(C_O), dim3(256), 0, stream>>>(conv_w, conv_b, bn_g, bn_b,
                                                    bn_m, bn_v, wp, scale, shift);
  prep_lin<<<dim3(64), dim3(256), 0, stream>>>(wq, wk, wvv, wo, wq16, wk16,
                                               wv16, wo16);
  conv_bn_elu<<<dim3(T_OP / CONV_NT, B_N), dim3(128), 0, stream>>>(
      x, wp, scale, shift, y32, y16, s16);
  qkv_proj<<<dim3(T_OP / 64, B_N), dim3(128), 0, stream>>>(s16, y16, wq16,
                                                           wk16, wv16, qT, kb, vb);
  attention<<<dim3(T_OP / 64, B_N), dim3(128), 0, stream>>>(qT, kb, vb, cx);
  out_proj<<<dim3(T_OP / 64, B_N), dim3(128), 0, stream>>>(cx, wo16, y32,
                                                           (float*)d_out);
}
